// TensorTrain_67568425501234
// MI455X (gfx1250) — matrix operand order repaired and checked by reading, not compiled, not run on hardware
//
#include <hip/hip_runtime.h>
#include <stdint.h>

typedef __attribute__((ext_vector_type(16))) __bf16 v16bf;
typedef __attribute__((ext_vector_type(8)))  float  v8f;

#define BSZ    16384
#define NCH    8
#define HDIM   256
#define RANK   16
#define NAGGR  4
#define TDIM   64
#define RCOL   272   // (RANK+1)*RANK
#define TILE_M 32    // two 16-row WMMA tiles per wave

// ws layout (uint32 units):
//  uPack  : [8 ch][4 a][16 nt][8 ks][32 lane][8 dw] = 1048576
//  utPack : [4 a][2 ks][32 lane][8 dw]              = 2048
//  uoPack : [4 a][16 nt][32 lane][8 dw]             = 16384
#define UTPACK_OFF (NCH*NAGGR*16*8*32*8)
#define UOPACK_OFF (UTPACK_OFF + NAGGR*2*32*8)

// byte strides inside uPack
#define KS_BYTES   1024                    // 32 lanes * 8 dw * 4B
#define NT_BYTES   (8 * KS_BYTES)          // 8192
#define CA_BYTES   (16 * NT_BYTES)         // 131072

// dynamic LDS partition (bytes)
#define SM_NH_STAGE 0         // 32*256 f32  = 32768
#define SM_NH_BF    32768     // 32*128 u32  = 16384
#define SM_RIS      49152     // 4*2*4096 u16 = 65536
#define SM_CARRY    114688    // 4*512 f32   = 8192
#define SM_TOTAL    122880

// round-half-up f32->bf16 pair pack: 2 adds + 1 v_perm_b32
__device__ __forceinline__ uint32_t pkbf_rh(float lo, float hi) {
  uint32_t x = __float_as_uint(lo) + 0x8000u;
  uint32_t y = __float_as_uint(hi) + 0x8000u;
  return __builtin_amdgcn_perm(y, x, 0x07060302u);  // {y[31:16], x[31:16]}
}
__device__ __forceinline__ unsigned short f2bf_rh(float f) {
  return (unsigned short)((__float_as_uint(f) + 0x8000u) >> 16);
}
__device__ __forceinline__ void unpack2(uint32_t u, float& lo, float& hi) {
  lo = __uint_as_float(u << 16);
  hi = __uint_as_float(u & 0xffff0000u);
}
__device__ __forceinline__ void wait_async0() {
#if __has_builtin(__builtin_amdgcn_s_wait_asynccnt)
  __builtin_amdgcn_s_wait_asynccnt(0);
#else
  asm volatile("s_wait_asynccnt 0" ::: "memory");
#endif
}

union Frag {
  uint32_t u[8];
  uint4    q[2];
  v16bf    v;
};

// ---- pack U[c][a][256][272] -> bf16 B-operand swizzle (cols 0..255 only) ----
__global__ void pack_U_kernel(const float* __restrict__ U, uint32_t* __restrict__ out) {
  int idx = blockIdx.x * blockDim.x + threadIdx.x;   // 1048576 threads
  int j    = idx & 7;
  int lane = (idx >> 3) & 31;
  int ks   = (idx >> 8) & 7;
  int nt   = (idx >> 11) & 15;
  int a    = (idx >> 15) & 3;
  int c    = (idx >> 17);
  int n = lane & 15, kh = lane >> 4;
  int k0  = ks * 32 + kh * 8 + 2 * j + 8 * (j >> 2);
  int col = nt * 16 + n;
  const float* base = U + (size_t)(c * NAGGR + a) * HDIM * RCOL;
  out[idx] = pkbf_rh(base[(size_t)k0 * RCOL + col], base[(size_t)(k0 + 1) * RCOL + col]);
}

// ---- pack U_type[4][64][16] ----
__global__ void pack_Ut_kernel(const float* __restrict__ Ut, uint32_t* __restrict__ out) {
  int idx = blockIdx.x * blockDim.x + threadIdx.x;   // 2048 threads
  int j    = idx & 7;
  int lane = (idx >> 3) & 31;
  int ks   = (idx >> 8) & 1;
  int a    = (idx >> 9) & 3;
  int n = lane & 15, kh = lane >> 4;
  int k0 = ks * 32 + kh * 8 + 2 * j + 8 * (j >> 2);
  out[idx] = pkbf_rh(Ut[(size_t)(a * TDIM + k0) * RANK + n],
                     Ut[(size_t)(a * TDIM + k0 + 1) * RANK + n]);
}

// ---- pack U_output[4][16][256] (K=16 zero-padded to 32) ----
__global__ void pack_Uo_kernel(const float* __restrict__ Uo, uint32_t* __restrict__ out) {
  int idx = blockIdx.x * blockDim.x + threadIdx.x;   // 16384 threads
  int j    = idx & 7;
  int lane = (idx >> 3) & 31;
  int nt   = (idx >> 8) & 15;
  int a    = (idx >> 12) & 3;
  int n = lane & 15, kh = lane >> 4;
  int k0 = kh * 8 + 2 * j + 8 * (j >> 2);
  float v0 = 0.f, v1 = 0.f;
  if (j < 4) {
    v0 = Uo[(size_t)(a * RANK + k0) * HDIM + nt * 16 + n];
    v1 = Uo[(size_t)(a * RANK + k0 + 1) * HDIM + nt * 16 + n];
  }
  out[idx] = pkbf_rh(v0, v1);
}

__global__ __launch_bounds__(128) void tt_main(
    const float* __restrict__ nh,      // [BS][8][256]
    const float* __restrict__ temb,    // [BS][64]
    const float* __restrict__ bvec,    // [8][4][1][272]
    const float* __restrict__ btype,   // [4][1][16]
    const float* __restrict__ bout,    // [4][1][256]
    const uint32_t* __restrict__ uPack,
    const uint32_t* __restrict__ utPack,
    const uint32_t* __restrict__ uoPack,
    float* __restrict__ out)           // [BS][1024]
{
  extern __shared__ char smem[];
  float*          nh_stage = (float*)(smem + SM_NH_STAGE);     // 32 rows x 256 f32
  uint32_t*       nh_bf    = (uint32_t*)(smem + SM_NH_BF);     // 32 rows x 128 pairs
  unsigned short* ris      = (unsigned short*)(smem + SM_RIS); // [a][t][16*256]
  float*          carry    = (float*)(smem + SM_CARRY);        // [a][t][16*16]

  const int tid  = threadIdx.x;
  const int lane = tid & 31;
  const int a    = tid >> 5;            // wave id == aggregator index
  const int row0 = blockIdx.x * TILE_M;
  const int m    = lane & 15;           // A-row / B-col / D-col
  const int kh   = lane >> 4;           // lane half
  const uint32_t laneOff = (uint32_t)lane * 32u;   // byte offset into a pack block

  unsigned short* risW = ris + a * 2 * 4096;   // two tiles per wave
  float*          carW = carry + a * 512;

  // ---------- carry0 = temb_tile @ U_type[a] + b_type[a] (both tiles) ----------
#pragma unroll
  for (int t = 0; t < 2; ++t) {
    v8f acc = {0.f, 0.f, 0.f, 0.f, 0.f, 0.f, 0.f, 0.f};
#pragma unroll
    for (int ks = 0; ks < 2; ++ks) {
      Frag A;
      const float* r1 = temb + (size_t)(row0 + t * 16 + m) * TDIM + ks * 32 + 8 * kh;
      const float* r2 = r1 + 16;
      float4 x0 = *(const float4*)r1, x1 = *(const float4*)(r1 + 4);
      float4 y0 = *(const float4*)r2, y1 = *(const float4*)(r2 + 4);
      A.u[0] = pkbf_rh(x0.x, x0.y); A.u[1] = pkbf_rh(x0.z, x0.w);
      A.u[2] = pkbf_rh(x1.x, x1.y); A.u[3] = pkbf_rh(x1.z, x1.w);
      A.u[4] = pkbf_rh(y0.x, y0.y); A.u[5] = pkbf_rh(y0.z, y0.w);
      A.u[6] = pkbf_rh(y1.x, y1.y); A.u[7] = pkbf_rh(y1.z, y1.w);
      Frag B;
      const char* bpb = (const char*)utPack + (size_t)(a * 2 + ks) * KS_BYTES + laneOff;
      B.q[0] = ((const uint4*)bpb)[0];
      B.q[1] = ((const uint4*)bpb)[1];
      acc = __builtin_amdgcn_wmma_f32_16x16x32_bf16(false, A.v, false, B.v,
                                                    (short)0, acc, false, false);
    }
    float bt = btype[a * RANK + m];
#pragma unroll
    for (int v = 0; v < 8; ++v)
      carW[t * 256 + (v + 8 * kh) * RANK + m] = acc[v] + bt;
  }

  // staging assignment: row sr (0..31), 64-float chunk sc0
  const int sr  = tid >> 2;
  const int sc0 = (tid & 3) * 64;
  const uint32_t lds_dst = (uint32_t)(uintptr_t)(nh_stage + sr * HDIM + sc0);

  // ---------- scan over channels ----------
  for (int c = 0; c < NCH; ++c) {
    __syncthreads();  // previous channel's nh readers done

    // async copy: global f32 tile -> nh_stage (16 x b128 per thread, ASYNCcnt)
    {
      const uint32_t goff = (uint32_t)((((row0 + sr) * NCH + c) * HDIM + sc0) * 4);
#pragma unroll
      for (int t = 0; t < 16; ++t) {
        asm volatile("global_load_async_to_lds_b128 %0, %1, %2 offset:%3"
                     :
                     : "v"(lds_dst), "v"(goff), "s"(nh), "n"(t * 16)
                     : "memory");
      }
      if (c + 1 < NCH)                  // gfx1250 global_prefetch_b8
        __builtin_prefetch(nh + ((size_t)(row0 + sr) * NCH + (c + 1)) * HDIM + sc0, 0, 1);
    }
    wait_async0();
    __syncthreads();

    // convert pass: f32 -> packed bf16 pairs (v_perm_b32 based)
    {
      const float* sp = nh_stage + sr * HDIM + sc0;
      uint32_t* dp = nh_bf + sr * (HDIM / 2) + (sc0 >> 1);
#pragma unroll
      for (int t = 0; t < 16; ++t) {
        float4 x = *(const float4*)(sp + 4 * t);
        dp[2 * t]     = pkbf_rh(x.x, x.y);
        dp[2 * t + 1] = pkbf_rh(x.z, x.w);
      }
    }
    __syncthreads();

    // A fragments for both tiles: pure ds_load_b128
    Frag Af[2][8];
#pragma unroll
    for (int t = 0; t < 2; ++t)
#pragma unroll
      for (int ks = 0; ks < 8; ++ks) {
        const uint32_t* ap = nh_bf + (t * 16 + m) * (HDIM / 2) + ks * 16 + 4 * kh;
        Af[t][ks].q[0] = *(const uint4*)ap;
        Af[t][ks].q[1] = *(const uint4*)(ap + 8);
      }

    // scalar base for this (c,a); vector part is laneOff only
    const char* ub_ca = (const char*)uPack + (size_t)(c * NAGGR + a) * CA_BYTES + laneOff;
    const float* bb_base = bvec + (size_t)(c * NAGGR + a) * RCOL;

    for (int nt = 0; nt < 16; ++nt) {
      const char* ub = ub_ca + nt * NT_BYTES;
      v8f d0 = {0.f, 0.f, 0.f, 0.f, 0.f, 0.f, 0.f, 0.f};
      v8f d1 = {0.f, 0.f, 0.f, 0.f, 0.f, 0.f, 0.f, 0.f};
#pragma unroll
      for (int ks = 0; ks < 8; ++ks) {
        Frag B;    // one B feeds both M-tiles: halves B traffic
        B.q[0] = *(const uint4*)(ub + ks * KS_BYTES);        // imm offset ks*1024
        B.q[1] = *(const uint4*)(ub + ks * KS_BYTES + 16);   // imm offset +16
        d0 = __builtin_amdgcn_wmma_f32_16x16x32_bf16(false, Af[0][ks].v, false, B.v,
                                                     (short)0, d0, false, false);
        d1 = __builtin_amdgcn_wmma_f32_16x16x32_bf16(false, Af[1][ks].v, false, B.v,
                                                     (short)0, d1, false, false);
      }
      float bb = bb_base[nt * 16 + m];
#pragma unroll
      for (int v = 0; v < 8; ++v) {
        risW[(v + 8 * kh) * HDIM + nt * 16 + m]        = f2bf_rh(d0[v] + bb);
        risW[4096 + (v + 8 * kh) * HDIM + nt * 16 + m] = f2bf_rh(d1[v] + bb);
      }
    }

    // per-row scan update for both tiles: carry = carry @ M + bias
#pragma unroll
    for (int t = 0; t < 2; ++t) {
      unsigned short* rp = risW + t * 4096;
      float*          cp = carW + t * 256;
      float cr[16];
      {
        const float4* c4 = (const float4*)(cp + m * RANK);
#pragma unroll
        for (int u = 0; u < 4; ++u) {
          float4 x = c4[u];
          cr[4*u] = x.x; cr[4*u+1] = x.y; cr[4*u+2] = x.z; cr[4*u+3] = x.w;
        }
      }
      float nv[8];
      {
        uint4 bq = *(const uint4*)&rp[m * HDIM + 240 + kh * 8];   // bias row 15
        unpack2(bq.x, nv[0], nv[1]); unpack2(bq.y, nv[2], nv[3]);
        unpack2(bq.z, nv[4], nv[5]); unpack2(bq.w, nv[6], nv[7]);
      }
#pragma unroll
      for (int r = 0; r < 16; ++r) {
        uint4 rq = *(const uint4*)&rp[m * HDIM + r * 16 + kh * 8];
        float crr = cr[r], x0, x1;
        unpack2(rq.x, x0, x1); nv[0] += crr * x0; nv[1] += crr * x1;
        unpack2(rq.y, x0, x1); nv[2] += crr * x0; nv[3] += crr * x1;
        unpack2(rq.z, x0, x1); nv[4] += crr * x0; nv[5] += crr * x1;
        unpack2(rq.w, x0, x1); nv[6] += crr * x0; nv[7] += crr * x1;
      }
      // wave-lockstep: all lanes' loads above issue before these stores
      float4* outp = (float4*)(cp + m * RANK + kh * 8);
      outp[0] = make_float4(nv[0], nv[1], nv[2], nv[3]);
      outp[1] = make_float4(nv[4], nv[5], nv[6], nv[7]);
    }
  }

  // ---------- out = carry @ U_output[a] + b_output[a] ----------
  Frag Afin[2];
#pragma unroll
  for (int t = 0; t < 2; ++t) {
    const float* cp = carW + t * 256;
#pragma unroll
    for (int j = 0; j < 4; ++j)
      Afin[t].u[j] = pkbf_rh(cp[m * RANK + 2 * j + 8 * kh],
                             cp[m * RANK + 2 * j + 1 + 8 * kh]);
    Afin[t].u[4] = Afin[t].u[5] = Afin[t].u[6] = Afin[t].u[7] = 0;
  }

  const char* uo_base = (const char*)uoPack + (size_t)a * (16 * KS_BYTES) + laneOff;
  for (int nt = 0; nt < 16; ++nt) {
    Frag B;
    B.q[0] = *(const uint4*)(uo_base + nt * KS_BYTES);
    B.q[1] = *(const uint4*)(uo_base + nt * KS_BYTES + 16);
    v8f d0 = {0.f, 0.f, 0.f, 0.f, 0.f, 0.f, 0.f, 0.f};
    v8f d1 = {0.f, 0.f, 0.f, 0.f, 0.f, 0.f, 0.f, 0.f};
    d0 = __builtin_amdgcn_wmma_f32_16x16x32_bf16(false, Afin[0].v, false, B.v,
                                                 (short)0, d0, false, false);
    d1 = __builtin_amdgcn_wmma_f32_16x16x32_bf16(false, Afin[1].v, false, B.v,
                                                 (short)0, d1, false, false);
    float bo = bout[a * HDIM + nt * 16 + m];
#pragma unroll
    for (int v = 0; v < 8; ++v) {
      // streaming output: non-temporal stores keep L2 for weights/nh
      __builtin_nontemporal_store(d0[v] + bo,
          &out[(size_t)(row0 + v + 8 * kh) * (NAGGR * HDIM) + a * HDIM + nt * 16 + m]);
      __builtin_nontemporal_store(d1[v] + bo,
          &out[(size_t)(row0 + 16 + v + 8 * kh) * (NAGGR * HDIM) + a * HDIM + nt * 16 + m]);
    }
  }
}

extern "C" void kernel_launch(void* const* d_in, const int* in_sizes, int n_in,
                              void* d_out, int out_size, void* d_ws, size_t ws_size,
                              hipStream_t stream) {
  (void)in_sizes; (void)n_in; (void)out_size; (void)ws_size;
  const float* nh   = (const float*)d_in[0];
  const float* temb = (const float*)d_in[1];
  const float* U    = (const float*)d_in[2];
  const float* bv   = (const float*)d_in[3];
  const float* Ut   = (const float*)d_in[4];
  const float* bt   = (const float*)d_in[5];
  const float* Uo   = (const float*)d_in[6];
  const float* bo   = (const float*)d_in[7];

  uint32_t* uPack  = (uint32_t*)d_ws;
  uint32_t* utPack = uPack + UTPACK_OFF;
  uint32_t* uoPack = uPack + UOPACK_OFF;

  pack_U_kernel <<<(NCH*NAGGR*16*8*32*8) / 256, 256, 0, stream>>>(U, uPack);
  pack_Ut_kernel<<<(NAGGR*2*32*8)        / 256, 256, 0, stream>>>(Ut, utPack);
  pack_Uo_kernel<<<(NAGGR*16*32*8)       / 256, 256, 0, stream>>>(Uo, uoPack);

  tt_main<<<BSZ / TILE_M, 128, SM_TOTAL, stream>>>(nh, temb, bv, bt, bo,
                                                   uPack, utPack, uoPack,
                                                   (float*)d_out);
}
